// POSHGNN_89378269430401
// MI455X (gfx1250) — compile-verified, matrix operand order checked by reading, not run on hardware
//
#include <hip/hip_runtime.h>
#include <cstdint>
#include <cstddef>

// ---------------------------------------------------------------------------
// Types for CDNA5 WMMA (gfx1250, wave32)
// ---------------------------------------------------------------------------
typedef __attribute__((ext_vector_type(16))) __bf16 v16bf;
typedef __attribute__((ext_vector_type(8)))  float  v8f;
typedef __attribute__((ext_vector_type(4)))  int    v4i;

#define DEVFN static __device__ __forceinline__

#if defined(__gfx1250__) && __has_builtin(__builtin_amdgcn_global_load_async_to_lds_b128)
#define HAVE_ASYNC_LDS 1
#else
#define HAVE_ASYNC_LDS 0
#endif

#if HAVE_ASYNC_LDS
typedef __attribute__((address_space(1))) v4i gas_v4i;   // global (addrspace 1)
typedef __attribute__((address_space(3))) v4i las_v4i;   // LDS (addrspace 3)
#endif

DEVFN void async_wait_all() {
#if HAVE_ASYNC_LDS
#if __has_builtin(__builtin_amdgcn_s_wait_asynccnt)
  __builtin_amdgcn_s_wait_asynccnt(0);
#else
  asm volatile("s_wait_asynccnt 0x0" ::: "memory");
#endif
#endif
}

// Copy one 16-byte chunk global -> LDS, async (DMA, no VGPR data) if available.
DEVFN void lds_copy16(__bf16* ldst, const __bf16* gsrc) {
#if HAVE_ASYNC_LDS
  __builtin_amdgcn_global_load_async_to_lds_b128(
      (gas_v4i*)gsrc, (las_v4i*)ldst, 0, 0);
#else
  *(uint4*)ldst = *(const uint4*)gsrc;
#endif
}

// Load one 16-lane-row fragment (A or B) for v_wmma_f32_16x16x32_bf16.
// rowptr = pointer to this lane's matrix row (row-major, any addrspace).
// Lane l, element e maps to K = (e>>3)*16 + ((l>>4)<<3) + (e&7)  (ISA 7.12.2),
// i.e. two contiguous 16-byte chunks per lane.
DEVFN v16bf load_frag_row(const __bf16* rowptr, int lane, int kb) {
  const int c0 = kb * 32 + ((lane >> 4) << 3);
  union { v16bf bf; uint4 q[2]; } f;
  f.q[0] = *(const uint4*)(rowptr + c0);
  f.q[1] = *(const uint4*)(rowptr + c0 + 16);
  return f.bf;
}

// ---------------------------------------------------------------------------
// Utility kernels
// ---------------------------------------------------------------------------
__global__ void k_fill_u32(unsigned* p, unsigned v, long long n) {
  long long i = (long long)blockIdx.x * blockDim.x + threadIdx.x;
  if (i < n) p[i] = v;
}

// Convert f32 weight (rows x cols, row-major) to bf16 padded to colsp columns.
__global__ void k_conv_pad(const float* __restrict__ Wf, __bf16* __restrict__ Wb,
                           int rows, int cols, int colsp) {
  int t = blockIdx.x * blockDim.x + threadIdx.x;
  if (t >= rows * colsp) return;
  int r = t / colsp, c = t - r * colsp;
  float v = (c < cols) ? Wf[(size_t)r * cols + c] : 0.f;
  Wb[t] = (__bf16)v;
}

// ---------------------------------------------------------------------------
// Pass 1: per-edge min/max/count aggregations (bit-pattern atomics, vals >= 0)
// ---------------------------------------------------------------------------
__global__ void k_edge_pre(const float* __restrict__ x,
                           const int* __restrict__ ei, const int* __restrict__ eip,
                           float* dint_out, float* int_out,
                           float* cnt1, float* cnt1p, int E) {
  int e = blockIdx.x * blockDim.x + threadIdx.x;
  if (e >= E) return;
  int s = ei[e], t = ei[E + e];
  float itf = x[(size_t)s * 4 + 2];
  float d   = x[(size_t)s * 4 + 3] + 1.f;
  float di  = d * itf;
  if (di == 0.f) di = 10000.f;
  atomicMin((unsigned*)&dint_out[t], __float_as_uint(di));
  atomicMax((unsigned*)&int_out[t],  __float_as_uint(itf));
  atomicAdd(&cnt1[t], 1.f);
  int tp = eip[E + e];
  atomicAdd(&cnt1p[tp], 1.f);
}

// Pass 2: o2 = segment_sum(o1[src]); o2p likewise.
__global__ void k_edge_deg2(const int* __restrict__ ei, const int* __restrict__ eip,
                            const float* __restrict__ cnt1, const float* __restrict__ cnt1p,
                            float* o2, float* o2p, int E) {
  int e = blockIdx.x * blockDim.x + threadIdx.x;
  if (e >= E) return;
  atomicAdd(&o2[ei[E + e]],  cnt1[ei[e]]);
  atomicAdd(&o2p[eip[E + e]], cnt1p[eip[e]]);
}

// ---------------------------------------------------------------------------
// Node pass: mask, ps, and build layer-0 feature rows (bf16, padded to 96)
// row = [ps0, ps1, 1, o1-o1p, o2-o2p, rec_prev, hidden_prev(64), 0 x26]
// ---------------------------------------------------------------------------
__global__ void k_node_feat0(const float* __restrict__ x,
                             const float* __restrict__ rec_prev,
                             const float* __restrict__ hid_prev,
                             const float* __restrict__ dint_out,
                             const float* __restrict__ int_out,
                             const float* __restrict__ cnt1,
                             const float* __restrict__ cnt1p,
                             const float* __restrict__ o2,
                             const float* __restrict__ o2p,
                             float* __restrict__ mask, float* __restrict__ ps,
                             __bf16* __restrict__ fb0, int N) {
  int i = blockIdx.x * blockDim.x + threadIdx.x;
  if (i >= N) return;
  float d   = x[(size_t)i * 4 + 3] + 1.f;
  float dio = dint_out[i];
  if (__float_as_uint(dio) == 0x7F800000u) dio = 0.f;   // no in-edges: min=+inf -> 0
  float io  = int_out[i];
  float m   = (d * io - dio <= 0.f) ? 1.f : 0.f;
  mask[i] = m;
  float inv = m / (d * d);
  float p0 = x[(size_t)i * 4 + 0] * inv;
  float p1 = x[(size_t)i * 4 + 1] * inv;
  ps[2 * i] = p0; ps[2 * i + 1] = p1;
  __bf16* row = fb0 + (size_t)i * 96;
  row[0] = (__bf16)p0;
  row[1] = (__bf16)p1;
  row[2] = (__bf16)1.f;
  row[3] = (__bf16)(cnt1[i] - cnt1p[i]);
  row[4] = (__bf16)(o2[i] - o2p[i]);
  row[5] = (__bf16)rec_prev[i];
  for (int c = 0; c < 64; ++c) row[6 + c] = (__bf16)hid_prev[(size_t)i * 64 + c];
  for (int c = 70; c < 96; ++c) row[c] = (__bf16)0.f;
}

// ---------------------------------------------------------------------------
// Edge feature-sum aggregation: sum[dst] += feat[src] (bf16 gather, f32 atomics)
// ---------------------------------------------------------------------------
template <int DP>
__global__ void k_agg_sum(const __bf16* __restrict__ feat, float* __restrict__ sum,
                          const int* __restrict__ ei, int E) {
  int e = blockIdx.x * blockDim.x + threadIdx.x;
  if (e >= E) return;
  int s = ei[e], d = ei[E + e];
  if (e + 512 < E)
    __builtin_prefetch(feat + (size_t)ei[e + 512] * DP, 0, 1);  // global_prefetch_b8
  const __bf16* fr = feat + (size_t)s * DP;
  float* so = sum + (size_t)d * DP;
#pragma unroll
  for (int c = 0; c < DP; c += 8) {
    uint4 q = *(const uint4*)(fr + c);
    const __bf16* b = (const __bf16*)&q;
#pragma unroll
    for (int j = 0; j < 8; ++j) atomicAdd(&so[c + j], (float)b[j]);
  }
}

// ps (2-wide) aggregation for pdr layer 0
__global__ void k_ps_agg(const float* __restrict__ ps, const int* __restrict__ ei,
                         float* __restrict__ aggps, int E) {
  int e = blockIdx.x * blockDim.x + threadIdx.x;
  if (e >= E) return;
  int s = ei[e], t = ei[E + e];
  atomicAdd(&aggps[2 * t],     ps[2 * s]);
  atomicAdd(&aggps[2 * t + 1], ps[2 * s + 1]);
}

// ---------------------------------------------------------------------------
// WMMA SAGE GEMM:  out[32 x 64] = mean[32 x DP] @ Wl^T + v[32 x DP] @ Wr^T + bl
// 256 threads = 8 waves; wave (grp,t) computes 16x16 tile (rows grp, cols t).
// Weights async-DMA'd into LDS (overlapped with mean staging); output relu'd,
// written bf16 as next layer's feature matrix.
// ---------------------------------------------------------------------------
template <int DP>
__global__ void k_gemm_sage(const float* __restrict__ sum, const float* __restrict__ cnt,
                            const __bf16* __restrict__ feat,
                            const __bf16* __restrict__ Wl, const __bf16* __restrict__ Wr,
                            const float* __restrict__ bl,
                            __bf16* __restrict__ fout, int N) {
  __shared__ __align__(16) __bf16 smean[32 * DP];
  __shared__ __align__(16) __bf16 swl[64 * DP];
  __shared__ __align__(16) __bf16 swr[64 * DP];
  __shared__ float sinv[32];
  const int row0 = blockIdx.x * 32;
  const int tid  = threadIdx.x;

  // 1) Kick off async global->LDS DMA of both weight matrices (ASYNCcnt).
  constexpr int WCHUNK = (64 * DP) / 8;   // 16B chunks per matrix
  for (int idx = tid; idx < WCHUNK; idx += 256) {
    lds_copy16(swl + idx * 8, Wl + idx * 8);
    lds_copy16(swr + idx * 8, Wr + idx * 8);
  }

  // 2) Per-row reciprocal degree.
  if (tid < 32) {
    int node = row0 + tid;
    if (node >= N) node = N - 1;
    sinv[tid] = 1.f / fmaxf(cnt[node], 1.f);
  }
  __syncthreads();

  // 3) Stage mean tile (sum * 1/cnt) into LDS as bf16; overlaps with the DMA.
  for (int idx = tid; idx < 32 * DP; idx += 256) {
    int r = idx / DP, c = idx - r * DP;
    int node = row0 + r;
    if (node >= N) node = N - 1;
    smean[idx] = (__bf16)(sum[(size_t)node * DP + c] * sinv[r]);
  }
  async_wait_all();        // each wave drains its own ASYNCcnt
  __syncthreads();         // publish LDS to all waves

  const int lane = tid & 31;
  const int w = tid >> 5, grp = w >> 2, t = w & 3;
  const int mrow = grp * 16 + (lane & 15);
  int vnode = row0 + mrow; if (vnode >= N) vnode = N - 1;

  const __bf16* rowA = smean + (size_t)mrow * DP;
  const __bf16* rowV = feat + (size_t)vnode * DP;
  const __bf16* rowL = swl + (size_t)(t * 16 + (lane & 15)) * DP;  // B[k][n]=W[n][k]
  const __bf16* rowR = swr + (size_t)(t * 16 + (lane & 15)) * DP;

  v8f acc = {0.f, 0.f, 0.f, 0.f, 0.f, 0.f, 0.f, 0.f};
#pragma unroll
  for (int kb = 0; kb < DP / 32; ++kb) {
    v16bf am = load_frag_row(rowA, lane, kb);
    v16bf wl = load_frag_row(rowL, lane, kb);
    acc = __builtin_amdgcn_wmma_f32_16x16x32_bf16(false, am, false, wl,
                                                  (short)0, acc, false, false);
    v16bf av = load_frag_row(rowV, lane, kb);
    v16bf wr = load_frag_row(rowR, lane, kb);
    acc = __builtin_amdgcn_wmma_f32_16x16x32_bf16(false, av, false, wr,
                                                  (short)0, acc, false, false);
  }

  // C/D layout: element g -> M = g + 8*(lane>>4), N = lane&15.
  const int col = t * 16 + (lane & 15);
  const float bias = bl[col];
#pragma unroll
  for (int g = 0; g < 8; ++g) {
    int m = g + ((lane >> 4) << 3);
    int node = row0 + grp * 16 + m;
    if (node < N) {
      float v = fmaxf(acc[g] + bias, 0.f);   // relu for next layer's input
      fout[(size_t)node * 64 + col] = (__bf16)v;
    }
  }
}

// ---------------------------------------------------------------------------
// dout=1 SAGE + sigmoid: out = sigmoid( mean.Wl + bl + v.Wr )
// ---------------------------------------------------------------------------
__global__ void k_sage_dot_sig(const float* __restrict__ sum, const float* __restrict__ cnt,
                               const __bf16* __restrict__ feat,
                               const float* __restrict__ Wl, const float* __restrict__ Wr,
                               const float* __restrict__ bl,
                               float* __restrict__ out, int N) {
  int i = blockIdx.x * blockDim.x + threadIdx.x;
  if (i >= N) return;
  float cn = fmaxf(cnt[i], 1.f);
  float a = bl[0];
  const float* srow = sum + (size_t)i * 64;
  const __bf16* frow = feat + (size_t)i * 64;
#pragma unroll 8
  for (int c = 0; c < 64; ++c)
    a += (srow[c] / cn) * Wl[c] + (float)frow[c] * Wr[c];
  out[i] = 1.f / (1.f + __expf(-a));
}

// pdr layer 0: h = mean(ps).Wl^T + bl + ps.Wr^T  (din=2, dout=64)
__global__ void k_pdr_h(const float* __restrict__ aggps, const float* __restrict__ cnt,
                        const float* __restrict__ ps,
                        const float* __restrict__ Wl, const float* __restrict__ Wr,
                        const float* __restrict__ bl,
                        float* __restrict__ hout, __bf16* __restrict__ fbh, long long NT) {
  long long t = (long long)blockIdx.x * blockDim.x + threadIdx.x;
  if (t >= NT) return;
  int i = (int)(t >> 6), j = (int)(t & 63);
  float cn = fmaxf(cnt[i], 1.f);
  float h = (aggps[2 * i] / cn) * Wl[j * 2] + (aggps[2 * i + 1] / cn) * Wl[j * 2 + 1]
          + bl[j] + ps[2 * i] * Wr[j * 2] + ps[2 * i + 1] * Wr[j * 2 + 1];
  hout[t] = h;                       // raw h -> d_out
  fbh[t] = (__bf16)fmaxf(h, 0.f);    // relu(h) -> bf16 features for next agg
}

// rec + block reductions (max(rec), sum(preserve))
__global__ void k_rec(const float* __restrict__ rec_prev, const float* __restrict__ pres,
                      const float* __restrict__ sel, const float* __restrict__ mask,
                      float* __restrict__ rec, unsigned* maxbits, float* sumpres, int N) {
  __shared__ float smax[256];
  __shared__ float ssum[256];
  int tid = threadIdx.x;
  int i = blockIdx.x * 256 + tid;
  float r = 0.f, p = 0.f;
  if (i < N) {
    p = pres[i];
    r = (rec_prev[i] * p + sel[i] * (1.f - p)) * mask[i];
    rec[i] = r;
  }
  smax[tid] = r; ssum[tid] = p;
  __syncthreads();
  for (int s = 128; s > 0; s >>= 1) {
    if (tid < s) { smax[tid] = fmaxf(smax[tid], smax[tid + s]); ssum[tid] += ssum[tid + s]; }
    __syncthreads();
  }
  if (tid == 0) {
    atomicMax(maxbits, __float_as_uint(smax[0]));   // rec >= 0 -> bit order valid
    atomicAdd(sumpres, ssum[0]);
  }
}

__global__ void k_finalize(const unsigned* maxbits, const float* sumpres,
                           float* scores_min, float* lwp_util, int N) {
  float mx = __uint_as_float(*maxbits) - 1e-7f;
  *scores_min = fminf(mx, 0.5f);
  *lwp_util = *sumpres / (float)N;
}

__global__ void k_perm(const float* __restrict__ rec, const float* __restrict__ scores_min,
                       float* __restrict__ perm, int N) {
  int i = blockIdx.x * blockDim.x + threadIdx.x;
  if (i < N) perm[i] = (rec[i] > *scores_min) ? 1.f : 0.f;
}

// ---------------------------------------------------------------------------
// Host launcher
// ---------------------------------------------------------------------------
struct SageW { const float *Wl, *Wr, *bl; };
static SageW pick_sage(void* const* d_in, const int* sz, int base, int dout, int din) {
  SageW s; s.Wl = (const float*)d_in[base];
  if (sz[base + 1] == dout * din) {            // order Wl, Wr, bl
    s.Wr = (const float*)d_in[base + 1]; s.bl = (const float*)d_in[base + 2];
  } else {                                     // order Wl, bl, Wr
    s.bl = (const float*)d_in[base + 1]; s.Wr = (const float*)d_in[base + 2];
  }
  return s;
}

extern "C" void kernel_launch(void* const* d_in, const int* in_sizes, int n_in,
                              void* d_out, int out_size, void* d_ws, size_t ws_size,
                              hipStream_t stream) {
  const int N = in_sizes[1];           // rec_prev: (N,1)
  const int E = in_sizes[3] / 2;       // edge_index: (2,E)
  const int TB = 256;
  const int gE = (E + TB - 1) / TB;
  const int gN = (N + TB - 1) / TB;

  const float* x        = (const float*)d_in[0];
  const float* rec_prev = (const float*)d_in[1];
  const float* hid_prev = (const float*)d_in[2];
  const int*   ei       = (const int*)d_in[3];
  const int*   eip      = (const int*)d_in[4];
  SageW L0 = pick_sage(d_in, in_sizes, 5, 64, 70);
  SageW L1 = pick_sage(d_in, in_sizes, 8, 64, 64);
  SageW L2 = pick_sage(d_in, in_sizes, 11, 64, 64);
  SageW L3 = pick_sage(d_in, in_sizes, 14, 1, 64);
  SageW P0 = pick_sage(d_in, in_sizes, 17, 64, 2);
  SageW P1 = pick_sage(d_in, in_sizes, 20, 1, 64);

  // workspace carve
  char* w = (char*)d_ws; size_t off = 0;
  auto take = [&](size_t b) { size_t o = off; off = (off + b + 255) & ~(size_t)255; return o; };
  float* dint   = (float*)(w + take((size_t)N * 4));
  float* intout = (float*)(w + take((size_t)N * 4));
  float* cnt1   = (float*)(w + take((size_t)N * 4));
  float* cnt1p  = (float*)(w + take((size_t)N * 4));
  float* o2     = (float*)(w + take((size_t)N * 4));
  float* o2p    = (float*)(w + take((size_t)N * 4));
  float* maskb  = (float*)(w + take((size_t)N * 4));
  float* ps     = (float*)(w + take((size_t)N * 8));
  float* aggps  = (float*)(w + take((size_t)N * 8));
  float* selb   = (float*)(w + take((size_t)N * 4));
  unsigned* scal = (unsigned*)(w + take(256));     // [0]=maxbits [1]=sumpres [2]=scores_min
  __bf16* wl0 = (__bf16*)(w + take(64 * 96 * 2));
  __bf16* wr0 = (__bf16*)(w + take(64 * 96 * 2));
  __bf16* wl1 = (__bf16*)(w + take(64 * 64 * 2));
  __bf16* wr1 = (__bf16*)(w + take(64 * 64 * 2));
  __bf16* wl2 = (__bf16*)(w + take(64 * 64 * 2));
  __bf16* wr2 = (__bf16*)(w + take(64 * 64 * 2));
  float*  sumb = (float*)(w + take((size_t)N * 96 * 4));
  __bf16* fb0 = (__bf16*)(w + take((size_t)N * 96 * 2));
  __bf16* fb1 = (__bf16*)(w + take((size_t)N * 64 * 2));
  __bf16* fb2 = (__bf16*)(w + take((size_t)N * 64 * 2));
  (void)ws_size; (void)n_in; (void)out_size;

  float* out_rec  = (float*)d_out;
  float* out_perm = out_rec + N;
  float* out_h    = out_perm + N;
  float* out_pres = out_h + (size_t)N * 64;
  float* out_util = out_pres + N;

  unsigned* maxbits = scal;
  float* sumpres = (float*)(scal + 1);
  float* smin    = (float*)(scal + 2);

  // -- init
  k_fill_u32<<<gN, TB, 0, stream>>>((unsigned*)dint, 0x7F800000u, N);      // +inf
  k_fill_u32<<<gN, TB, 0, stream>>>((unsigned*)intout, 0u, N);
  k_fill_u32<<<gN, TB, 0, stream>>>((unsigned*)cnt1, 0u, N);
  k_fill_u32<<<gN, TB, 0, stream>>>((unsigned*)cnt1p, 0u, N);
  k_fill_u32<<<gN, TB, 0, stream>>>((unsigned*)o2, 0u, N);
  k_fill_u32<<<gN, TB, 0, stream>>>((unsigned*)o2p, 0u, N);
  k_fill_u32<<<(2 * N + TB - 1) / TB, TB, 0, stream>>>((unsigned*)aggps, 0u, 2LL * N);
  k_fill_u32<<<1, 64, 0, stream>>>(scal, 0u, 64);

  // -- weight conversion to bf16 (padded)
  k_conv_pad<<<(64 * 96 + TB - 1) / TB, TB, 0, stream>>>(L0.Wl, wl0, 64, 70, 96);
  k_conv_pad<<<(64 * 96 + TB - 1) / TB, TB, 0, stream>>>(L0.Wr, wr0, 64, 70, 96);
  k_conv_pad<<<(64 * 64 + TB - 1) / TB, TB, 0, stream>>>(L1.Wl, wl1, 64, 64, 64);
  k_conv_pad<<<(64 * 64 + TB - 1) / TB, TB, 0, stream>>>(L1.Wr, wr1, 64, 64, 64);
  k_conv_pad<<<(64 * 64 + TB - 1) / TB, TB, 0, stream>>>(L2.Wl, wl2, 64, 64, 64);
  k_conv_pad<<<(64 * 64 + TB - 1) / TB, TB, 0, stream>>>(L2.Wr, wr2, 64, 64, 64);

  // -- graph pre-passes
  k_edge_pre<<<gE, TB, 0, stream>>>(x, ei, eip, dint, intout, cnt1, cnt1p, E);
  k_edge_deg2<<<gE, TB, 0, stream>>>(ei, eip, cnt1, cnt1p, o2, o2p, E);
  k_node_feat0<<<gN, TB, 0, stream>>>(x, rec_prev, hid_prev, dint, intout,
                                      cnt1, cnt1p, o2, o2p, maskb, ps, fb0, N);

  const int gRows = (N + 31) / 32;
  const long long n96 = 96LL * N, n64 = 64LL * N;
  const int g96 = (int)((n96 + TB - 1) / TB), g64 = (int)((n64 + TB - 1) / TB);

  // -- lwp layer 0 (din 70->96 padded)
  k_fill_u32<<<g96, TB, 0, stream>>>((unsigned*)sumb, 0u, n96);
  k_agg_sum<96><<<gE, TB, 0, stream>>>(fb0, sumb, ei, E);
  k_gemm_sage<96><<<gRows, TB, 0, stream>>>(sumb, cnt1, fb0, wl0, wr0, L0.bl, fb1, N);

  // -- lwp layer 1
  k_fill_u32<<<g64, TB, 0, stream>>>((unsigned*)sumb, 0u, n64);
  k_agg_sum<64><<<gE, TB, 0, stream>>>(fb1, sumb, ei, E);
  k_gemm_sage<64><<<gRows, TB, 0, stream>>>(sumb, cnt1, fb1, wl1, wr1, L1.bl, fb2, N);

  // -- lwp layer 2
  k_fill_u32<<<g64, TB, 0, stream>>>((unsigned*)sumb, 0u, n64);
  k_agg_sum<64><<<gE, TB, 0, stream>>>(fb2, sumb, ei, E);
  k_gemm_sage<64><<<gRows, TB, 0, stream>>>(sumb, cnt1, fb2, wl2, wr2, L2.bl, fb1, N);

  // -- lwp layer 3 (dout=1) -> preserve (sigmoid), written straight into d_out
  k_fill_u32<<<g64, TB, 0, stream>>>((unsigned*)sumb, 0u, n64);
  k_agg_sum<64><<<gE, TB, 0, stream>>>(fb1, sumb, ei, E);
  k_sage_dot_sig<<<gN, TB, 0, stream>>>(sumb, cnt1, fb1, L3.Wl, L3.Wr, L3.bl, out_pres, N);

  // -- pdr layer 0: h (raw -> d_out, relu bf16 -> fb2)
  k_ps_agg<<<gE, TB, 0, stream>>>(ps, ei, aggps, E);
  k_pdr_h<<<g64, TB, 0, stream>>>(aggps, cnt1, ps, P0.Wl, P0.Wr, P0.bl, out_h, fb2, n64);

  // -- pdr layer 1 -> sel (sigmoid)
  k_fill_u32<<<g64, TB, 0, stream>>>((unsigned*)sumb, 0u, n64);
  k_agg_sum<64><<<gE, TB, 0, stream>>>(fb2, sumb, ei, E);
  k_sage_dot_sig<<<gN, TB, 0, stream>>>(sumb, cnt1, fb2, P1.Wl, P1.Wr, P1.bl, selb, N);

  // -- rec, reductions, perm, lwp_util
  k_rec<<<gN, TB, 0, stream>>>(rec_prev, out_pres, selb, maskb, out_rec, maxbits, sumpres, N);
  k_finalize<<<1, 1, 0, stream>>>(maxbits, sumpres, smin, out_util, N);
  k_perm<<<gN, TB, 0, stream>>>(out_rec, smin, out_perm, N);
}
